// MSMLTransformerLayer_86500641342117
// MI455X (gfx1250) — matrix operand order repaired and checked by reading, not compiled, not run on hardware
//
#include <hip/hip_runtime.h>
#include <hip/hip_bf16.h>

// ---------------------------------------------------------------------------
// MSML transformer layer for MI455X (gfx1250), wave32 + WMMA f16 16x16x32.
// - Exploits the att[:, :N] slice: only windows w<16 matter.
// - Every WMMA fragment load in every GEMM is a contiguous 128-bit load:
//   producers write operands in consumer-fragment layout (transposes fused
//   into epilogue stores, which were scalar anyway).
// ---------------------------------------------------------------------------

#define BB   2
#define NN   4096
#define DD   256
#define WINW 16
#define NW   256           // N / WIN
#define TOK  (BB * NN)     // 8192 tokens

typedef _Float16 v16h __attribute__((ext_vector_type(16)));
typedef _Float16 v8h  __attribute__((ext_vector_type(8)));
typedef float    v8f  __attribute__((ext_vector_type(8)));

__device__ __forceinline__ float silu_f(float x) { return x / (1.0f + __expf(-x)); }
// phi(z) = elu(elu(z)+1)+1 with elu(z)+1 > 0  ==>  z>0 ? z+2 : exp(z)+1
__device__ __forceinline__ float phi_f(float z)  { return z > 0.0f ? z + 2.0f : __expf(z) + 1.0f; }

// --- WMMA fragment loaders (row-major f16 source, ld in elements) ----------
// A fragment 16x32: lane L<16 -> row L, K {k0..k0+7, k0+16..k0+23};
//                   lane L>=16 -> row L-16, K {k0+8..15, k0+24..31}
__device__ __forceinline__ v16h load_a_frag(const _Float16* A, int ld, int k0, int lane) {
    int r = lane & 15, hi = lane >> 4;
    const _Float16* p = A + (size_t)r * ld + k0 + hi * 8;
    v8h lo = *(const v8h*)p;
    v8h hc = *(const v8h*)(p + 16);
    v16h o;
#pragma unroll
    for (int i = 0; i < 8; ++i) { o[i] = lo[i]; o[i + 8] = hc[i]; }
    return o;
}

// B fragment 32x16 = (row-major W[n,k])^T: lane L<16 -> col L, K {k0..k0+7, k0+16..k0+23};
//                    lane L>=16 -> col L-16, K {k0+8..15, k0+24..31}
__device__ __forceinline__ v16h load_b_frag(const _Float16* Wm, int ld, int k0, int lane) {
    int c = lane & 15, hi = lane >> 4;
    const _Float16* p = Wm + (size_t)c * ld + k0 + hi * 8;
    v8h x0 = *(const v8h*)p;
    v8h x1 = *(const v8h*)(p + 16);
    v16h o;
#pragma unroll
    for (int i = 0; i < 8; ++i) { o[i] = x0[i]; o[i + 8] = x1[i]; }
    return o;
}

__device__ __forceinline__ v8f wmma_f16(v16h a, v16h b, v8f c) {
    return __builtin_amdgcn_wmma_f32_16x16x32_f16(false, a, false, b, (short)0, c, false, false);
}

// --- small prep kernels ----------------------------------------------------
__global__ void cvt_f32_f16(const float* __restrict__ s, _Float16* __restrict__ d, int n) {
    int i = blockIdx.x * blockDim.x + threadIdx.x;
    if (i < n) d[i] = (_Float16)s[i];
}

// p_buf[(b*256+wn)*16+m] = embed_qk[1][b][w][n][m][:]·p_w + p_b   (w<16 only)
__global__ void pbias_kernel(const float* __restrict__ e, const float* __restrict__ pw,
                             const float* __restrict__ pb, float* __restrict__ pbuf) {
    int i = blockIdx.x * blockDim.x + threadIdx.x;      // B*256*16 = 8192
    if (i >= BB * 256 * 16) return;
    int m = i & 15, wn = (i >> 4) & 255, b = i >> 12;
    int w = wn >> 4, n = wn & 15;
    size_t off = ((((size_t)(1 * BB + b) * NW + w) * WINW + n) * WINW + m) * 3;
    pbuf[i] = e[off] * pw[0] + e[off + 1] * pw[1] + e[off + 2] * pw[2] + pb[0];
}

// LN over D=256, one token per 256-thread block; emits f16 x and f16 LN(x)
__global__ void ln_x_kernel(const float* __restrict__ x, const float* __restrict__ g,
                            const float* __restrict__ bt, _Float16* __restrict__ xh,
                            _Float16* __restrict__ hlnh) {
    int t = blockIdx.x, i = threadIdx.x;
    size_t idx = (size_t)t * DD + i;
    float v = x[idx];
    __shared__ float red[DD];
    red[i] = v; __syncthreads();
#pragma unroll
    for (int s = 128; s; s >>= 1) { if (i < s) red[i] += red[i + s]; __syncthreads(); }
    float mean = red[0] * (1.0f / DD); __syncthreads();
    float d = v - mean;
    red[i] = d * d; __syncthreads();
#pragma unroll
    for (int s = 128; s; s >>= 1) { if (i < s) red[i] += red[i + s]; __syncthreads(); }
    float var = red[0] * (1.0f / DD);
    float nv = d * rsqrtf(var + 1e-5f) * g[i] + bt[i];
    xh[idx] = (_Float16)v;
    hlnh[idx] = (_Float16)nv;
}

__global__ void ln_out_kernel(const float* __restrict__ hh, const float* __restrict__ g,
                              const float* __restrict__ bt, float* __restrict__ out) {
    int t = blockIdx.x, i = threadIdx.x;
    size_t idx = (size_t)t * DD + i;
    float v = hh[idx];
    __shared__ float red[DD];
    red[i] = v; __syncthreads();
#pragma unroll
    for (int s = 128; s; s >>= 1) { if (i < s) red[i] += red[i + s]; __syncthreads(); }
    float mean = red[0] * (1.0f / DD); __syncthreads();
    float d = v - mean;
    red[i] = d * d; __syncthreads();
#pragma unroll
    for (int s = 128; s; s >>= 1) { if (i < s) red[i] += red[i + s]; __syncthreads(); }
    float var = red[0] * (1.0f / DD);
    out[idx] = d * rsqrtf(var + 1e-5f) * g[i] + bt[i];
}

// --- token-level GEMM: out = epilogue(A(Mx256) @ W(256x256)^T + bias) ------
// block = 128 (4 waves); wave j owns columns [j*64, j*64+64): 4 B-frags per A-frag.
// mode 0: out32 = silu(v)
// mode 1: out32 = silu(v); out16 = (f16)silu(v); out16t[bm][col][w] = (f16)silu(v)
// mode 2: out16 = (f16)phi(v)                       (q: normal layout)
// mode 3: out32 = v + res[idx]                      (lin projection + residual h)
// mode 4: out16t[bm][col][w] = (f16)phi(v)          (k: window-transposed only)
// Transposed layout: index ((b*16+m)*256 + col)*256 + w, with token = row,
// b = row>>12, w = (row&4095)>>4, m = row&15.  It is exactly the fragment
// layout the KV GEMM wants (A: rows=c contiguous-k, B: rows=d contiguous-k).
__global__ void gemm_tok(const _Float16* __restrict__ A, const _Float16* __restrict__ Wm,
                         const float* __restrict__ bias, const float* __restrict__ res,
                         float* __restrict__ out32, _Float16* __restrict__ out16,
                         _Float16* __restrict__ out16t, int mode) {
    int lane = threadIdx.x & 31, wave = threadIdx.x >> 5;
    int hi = lane >> 4, lr = lane & 15;
    int row0 = blockIdx.x * 16;
    int n0 = wave * 64;
    const _Float16* Ab = A + (size_t)row0 * DD;
    v8f acc[4] = {v8f{}, v8f{}, v8f{}, v8f{}};
#pragma unroll
    for (int k0 = 0; k0 < DD; k0 += 32) {
        v16h a = load_a_frag(Ab, DD, k0, lane);
#pragma unroll
        for (int j = 0; j < 4; ++j) {
            v16h b = load_b_frag(Wm + (size_t)(n0 + j * 16) * DD, DD, k0, lane);
            acc[j] = wmma_f16(a, b, acc[j]);
        }
    }
    // tile-invariant transposed-coordinate pieces
    int bb = row0 >> 12;            // batch (full-M launches only)
    int ww = (row0 & 4095) >> 4;    // window index; all 16 rows share it
#pragma unroll
    for (int j = 0; j < 4; ++j) {
        int col = n0 + j * 16 + lr;
#pragma unroll
        for (int v = 0; v < 8; ++v) {
            int row = row0 + v + 8 * hi;
            size_t idx = (size_t)row * DD + col;
            float val = acc[j][v] + bias[col];
            if (mode == 0) {
                out32[idx] = silu_f(val);
            } else if (mode == 1) {
                float s = silu_f(val);
                out32[idx] = s;
                out16[idx] = (_Float16)s;
                int m = (v + 8 * hi);   // row & 15
                out16t[(((size_t)(bb * 16 + m) * DD) + col) * DD + ww] = (_Float16)s;
            } else if (mode == 2) {
                out16[idx] = (_Float16)phi_f(val);
            } else if (mode == 3) {
                out32[idx] = val + res[idx];
            } else {
                int m = (v + 8 * hi);
                out16t[(((size_t)(bb * 16 + m) * DD) + col) * DD + ww] = (_Float16)phi_f(val);
            }
        }
    }
}

// --- KV[b,m] = phi_k[b,:,m,:]^T @ v[b,:,m,:]   (256x256, K = W = 256) ------
// Operands pre-transposed: phikT[bm][c][w], vT[bm][d][w]  (ld = 256 over w).
// Output written transposed for the att GEMM: KVT[bm][d][c].
// grid.x = 16 c-tiles, grid.y = 32 (b,m); block = 128 (wave j: d in [j*64,..)).
__global__ void kv_kernel(const _Float16* __restrict__ phikT, const _Float16* __restrict__ vT,
                          _Float16* __restrict__ kvT) {
    int lane = threadIdx.x & 31, wave = threadIdx.x >> 5;
    int hi = lane >> 4, lr = lane & 15;
    int bm = blockIdx.y;
    int c0 = blockIdx.x * 16, d0 = wave * 64;
    const _Float16* Ab = phikT + (size_t)bm * DD * DD + (size_t)c0 * DD;
    const _Float16* Bb = vT + (size_t)bm * DD * DD;
    v8f acc[4] = {v8f{}, v8f{}, v8f{}, v8f{}};
#pragma unroll
    for (int w0 = 0; w0 < NW; w0 += 32) {
        v16h a = load_a_frag(Ab, DD, w0, lane);
#pragma unroll
        for (int j = 0; j < 4; ++j) {
            v16h bf = load_b_frag(Bb + (size_t)(d0 + j * 16) * DD, DD, w0, lane);
            acc[j] = wmma_f16(a, bf, acc[j]);
        }
    }
    // D element (c = c0 + v + 8*hi, d): 8 consecutive c per lane -> b128 store
#pragma unroll
    for (int j = 0; j < 4; ++j) {
        int d = d0 + j * 16 + lr;
        v8h st;
#pragma unroll
        for (int v = 0; v < 8; ++v) st[v] = (_Float16)acc[j][v];
        *(v8h*)(kvT + (size_t)bm * DD * DD + (size_t)d * DD + c0 + 8 * hi) = st;
    }
}

// --- att + bias + gate:  U[b, wn*16+m, d] = (phiq[b,wn,:]·KV[b,m][:,d] + p) * act_res
// B operand from KVT[bm][d][c] (contiguous in k=c).
// grid.x = 16 wn-tiles, grid.y = 32 (b,m); block = 128.
__global__ void att_kernel(const _Float16* __restrict__ phiq, const _Float16* __restrict__ kvT,
                           const float* __restrict__ pbuf, const float* __restrict__ actres,
                           _Float16* __restrict__ U) {
    int lane = threadIdx.x & 31, wave = threadIdx.x >> 5;
    int hi = lane >> 4, lr = lane & 15;
    int bm = blockIdx.y, b = bm >> 4, m = bm & 15;
    int wn0 = blockIdx.x * 16, d0 = wave * 64;
    const _Float16* Ab = phiq + (size_t)b * 256 * DD + (size_t)wn0 * DD;
    const _Float16* Kb = kvT + (size_t)bm * DD * DD;
    v8f acc[4] = {v8f{}, v8f{}, v8f{}, v8f{}};
#pragma unroll
    for (int k0 = 0; k0 < DD; k0 += 32) {
        v16h a = load_a_frag(Ab, DD, k0, lane);
#pragma unroll
        for (int j = 0; j < 4; ++j) {
            v16h bf = load_b_frag(Kb + (size_t)(d0 + j * 16) * DD, DD, k0, lane);
            acc[j] = wmma_f16(a, bf, acc[j]);
        }
    }
#pragma unroll
    for (int j = 0; j < 4; ++j) {
        int d = d0 + j * 16 + lr;
#pragma unroll
        for (int v = 0; v < 8; ++v) {
            int wn = wn0 + v + 8 * hi;
            int t = wn * 16 + m;                 // < 4096 since wn < 256
            size_t idx = ((size_t)b * NN + t) * DD + d;
            float val = acc[j][v] + pbuf[((size_t)b * 256 + wn) * 16 + m];
            U[idx] = (_Float16)(val * actres[idx]);
        }
    }
}

// ---------------------------------------------------------------------------
extern "C" void kernel_launch(void* const* d_in, const int* in_sizes, int n_in,
                              void* d_out, int out_size, void* d_ws, size_t ws_size,
                              hipStream_t stream) {
    const float* x        = (const float*)d_in[0];
    const float* embed    = (const float*)d_in[1];
    const float* act_w    = (const float*)d_in[2];
    const float* act_b    = (const float*)d_in[3];
    const float* pln_g    = (const float*)d_in[4];
    const float* pln_b    = (const float*)d_in[5];
    const float* conv_w   = (const float*)d_in[6];
    const float* conv_b   = (const float*)d_in[7];
    const float* q_w      = (const float*)d_in[8];
    const float* q_b      = (const float*)d_in[9];
    const float* k_w      = (const float*)d_in[10];
    const float* k_b      = (const float*)d_in[11];
    const float* p_w      = (const float*)d_in[12];
    const float* p_b      = (const float*)d_in[13];
    const float* lin_w    = (const float*)d_in[14];
    const float* lin_b    = (const float*)d_in[15];
    const float* norm_g   = (const float*)d_in[16];
    const float* norm_b   = (const float*)d_in[17];
    float* out = (float*)d_out;

    // workspace carve-out (256B aligned)
    char* base = (char*)d_ws;
    size_t off = 0;
    auto carve = [&](size_t bytes) { char* p = base + off; off += (bytes + 255) & ~(size_t)255; return p; };
    _Float16* xh     = (_Float16*)carve((size_t)TOK * DD * 2);
    _Float16* hlnh   = (_Float16*)carve((size_t)TOK * DD * 2);
    _Float16* h16    = (_Float16*)carve((size_t)TOK * DD * 2);   // token-major h (GEMM A input)
    _Float16* vT     = (_Float16*)carve((size_t)TOK * DD * 2);   // h transposed: [bm][d][w]
    _Float16* phikT  = (_Float16*)carve((size_t)TOK * DD * 2);   // phi_k transposed: [bm][c][w]
    _Float16* phiq   = (_Float16*)carve((size_t)BB * 256 * DD * 2);
    _Float16* wa16   = (_Float16*)carve((size_t)DD * DD * 2);
    _Float16* wc16   = (_Float16*)carve((size_t)DD * DD * 2);
    _Float16* wq16   = (_Float16*)carve((size_t)DD * DD * 2);
    _Float16* wk16   = (_Float16*)carve((size_t)DD * DD * 2);
    _Float16* wl16   = (_Float16*)carve((size_t)DD * DD * 2);
    _Float16* kvT    = (_Float16*)carve((size_t)BB * 16 * DD * DD * 2);  // [bm][d][c]
    _Float16* U16    = (_Float16*)carve((size_t)TOK * DD * 2);
    float*    actres = (float*)carve((size_t)TOK * DD * 4);
    float*    h32    = (float*)carve((size_t)TOK * DD * 4);
    float*    hidh   = (float*)carve((size_t)TOK * DD * 4);
    float*    pbuf   = (float*)carve((size_t)BB * 256 * 16 * 4);
    (void)ws_size; (void)in_sizes; (void)n_in; (void)out_size;

    const int WN = DD * DD;                       // 65536
    // 1) weight conversion + position bias
    cvt_f32_f16<<<WN / 256, 256, 0, stream>>>(act_w,  wa16, WN);
    cvt_f32_f16<<<WN / 256, 256, 0, stream>>>(conv_w, wc16, WN);
    cvt_f32_f16<<<WN / 256, 256, 0, stream>>>(q_w,    wq16, WN);
    cvt_f32_f16<<<WN / 256, 256, 0, stream>>>(k_w,    wk16, WN);
    cvt_f32_f16<<<WN / 256, 256, 0, stream>>>(lin_w,  wl16, WN);
    pbias_kernel<<<(BB * 256 * 16) / 256, 256, 0, stream>>>(embed, p_w, p_b, pbuf);

    // 2) LayerNorm of x -> xh (f16 copy of x) and hlnh (f16 LN(x))
    ln_x_kernel<<<TOK, DD, 0, stream>>>(x, pln_g, pln_b, xh, hlnh);

    // 3) act_res = silu(x @ Wa^T + ba)                             [mode 0]
    gemm_tok<<<TOK / 16, 128, 0, stream>>>(xh, wa16, act_b, nullptr, actres, nullptr, nullptr, 0);
    // 4) h = silu(LN(x) @ Wc^T + bc) -> f32 + f16 + transposed v   [mode 1]
    gemm_tok<<<TOK / 16, 128, 0, stream>>>(hlnh, wc16, conv_b, nullptr, h32, h16, vT, 1);
    // 5) phi_k = phi(h @ Wk^T + bk), written transposed            [mode 4]
    gemm_tok<<<TOK / 16, 128, 0, stream>>>(h16, wk16, k_b, nullptr, nullptr, nullptr, phikT, 4);
    // 6) phi_q, only tokens 0..255 per batch                       [mode 2]
    for (int b = 0; b < BB; ++b)
        gemm_tok<<<256 / 16, 128, 0, stream>>>(h16 + (size_t)b * NN * DD, wq16, q_b, nullptr,
                                               nullptr, phiq + (size_t)b * 256 * DD, nullptr, 2);
    // 7) KV[b,m] = phi_k^T @ v over all 256 windows (all-b128 fragment loads)
    kv_kernel<<<dim3(16, 32), 128, 0, stream>>>(phikT, vT, kvT);
    // 8) att + position bias, gated by act_res -> f16 GEMM input U
    att_kernel<<<dim3(16, 32), 128, 0, stream>>>(phiq, kvT, pbuf, actres, U16);
    // 9) hid + h = U @ Wl^T + bl + h                               [mode 3]
    gemm_tok<<<TOK / 16, 128, 0, stream>>>(U16, wl16, lin_b, h32, hidh, nullptr, nullptr, 3);
    // 10) out = LN(hid + h)
    ln_out_kernel<<<TOK, DD, 0, stream>>>(hidh, norm_g, norm_b, out);
}
